// _AttentionBlock_83829171683909
// MI455X (gfx1250) — compile-verified, matrix operand order checked by reading, not run on hardware
//
#include <hip/hip_runtime.h>
#include <hip/hip_bf16.h>
#include <math.h>

typedef __attribute__((ext_vector_type(16))) _Float16 v16h;
typedef __attribute__((ext_vector_type(8)))  _Float16 v8h;
typedef __attribute__((ext_vector_type(8)))  float    v8f;

constexpr int Bc  = 2;
constexpr int Sc  = 2048;
constexpr int Dc  = 1024;
constexpr int Hc  = 16;
constexpr int HDc = 64;
constexpr int DFFc = 4096;
constexpr int Mrows = Bc * Sc;   // 4096

// padded LDS row stride for a 32-half tile row: 32 + 8 pad halfs = 80 bytes
// -> (20*r) mod 64 banks distinct for r=0..15: conflict-free fragment reads
constexpr int TSH = 40;

// ---------------------------------------------------------------------------
// Fragment loaders. CDNA5 16-bit A-operand layout (ISA 7.12.2): lanes 0-15
// hold row `row`, K-chunks {0..7, 16..23}; lanes 16-31 same row, K-chunks
// {8..15, 24..31}. B-operand uses the same pattern on B^T.
// ---------------------------------------------------------------------------
__device__ __forceinline__ v16h load_frag(const _Float16* __restrict__ base,
                                          int row, int ld, int k0, int half) {
  const _Float16* p = base + (size_t)row * ld + k0 + 8 * half;
  v8h lo = *(const v8h*)(p);
  v8h hi = *(const v8h*)(p + 16);
  v16h r;
#pragma unroll
  for (int i = 0; i < 8; ++i) { r[i] = lo[i]; r[i + 8] = hi[i]; }
  return r;
}

__device__ __forceinline__ v16h load_frag_lds(const _Float16* base, int row, int half) {
  const _Float16* p = base + row * TSH + 8 * half;
  v8h lo = *(const v8h*)(p);
  v8h hi = *(const v8h*)(p + 16);
  v16h r;
#pragma unroll
  for (int i = 0; i < 8; ++i) { r[i] = lo[i]; r[i + 8] = hi[i]; }
  return r;
}

// low 32 bits of a flat pointer into LDS = workgroup-relative LDS byte offset
__device__ __forceinline__ unsigned lds_off(const void* p) {
  return (unsigned)(uintptr_t)p;
}

// ---------------------------------------------------------------------------
// Tensor Data Mover: DMA a 128-row x 32-half tile (row stride ldK halfs in
// global) into LDS with 16-byte padding after each 64-byte row.
// ---------------------------------------------------------------------------
#if __has_builtin(__builtin_amdgcn_tensor_load_to_lds)
#define HAVE_TDM 1
typedef __attribute__((ext_vector_type(4))) unsigned int u32x4;
typedef __attribute__((ext_vector_type(8))) int i32x8;
typedef __attribute__((ext_vector_type(4))) int i32x4;

__device__ __forceinline__ void tdm_load_tile32(const _Float16* gsrc, int ldK,
                                                unsigned lds_byte) {
  const unsigned long long ga = (unsigned long long)(uintptr_t)gsrc;
  u32x4 g0;
  g0[0] = 1u;                                                // count=1
  g0[1] = lds_byte;                                          // lds_addr
  g0[2] = (unsigned)ga;                                      // global_addr[31:0]
  g0[3] = (unsigned)((ga >> 32) & 0x01FFFFFFull) | 0x80000000u; // [56:32] | type=2
  i32x8 g1;
  g1[0] = (1 << 16)      // data_size = 2 bytes
        | (1 << 20)      // pad_enable
        | (3 << 22)      // pad_interval: every 16 dwords (=64B row)
        | (3 << 25);     // pad_amount: 4 dwords (=16B)
  g1[1] = (32 << 16);    // tensor_dim0 = 32 (tile-as-tensor: no OOB)
  g1[2] = (128 << 16);   // tensor_dim1 = 128
  g1[3] = (32 << 16);    // tile_dim0 = 32
  g1[4] = 128;           // tile_dim1 = 128, tile_dim2 = 0
  g1[5] = ldK;           // tensor_dim0_stride low32 (halfs)
  g1[6] = 0;             // stride hi | dim1_stride lo
  g1[7] = 0;
  i32x4 z4 = {0, 0, 0, 0};
#if defined(__clang_major__) && (__clang_major__ >= 23)
  i32x8 z8 = {0, 0, 0, 0, 0, 0, 0, 0};
  __builtin_amdgcn_tensor_load_to_lds(g0, g1, z4, z4, z8, 0);
#else
  __builtin_amdgcn_tensor_load_to_lds(g0, g1, z4, z4, 0);
#endif
}

__device__ __forceinline__ void wait_tensor0() {
#if __has_builtin(__builtin_amdgcn_s_wait_tensorcnt)
  __builtin_amdgcn_s_wait_tensorcnt(0);
#else
  asm volatile("s_wait_tensorcnt 0x0" ::: "memory");
#endif
}
#else
#define HAVE_TDM 0
// synchronous fallback: 256 threads stage both 128x32 tiles (padded layout)
__device__ __forceinline__ void stage_sync(_Float16* dstA, const _Float16* gA,
                                           _Float16* dstB, const _Float16* gB,
                                           int ldK) {
  const int t = threadIdx.x;
  const int row = t & 127;
  const _Float16* s = (t < 128 ? gA : gB) + (size_t)row * ldK;
  _Float16* d = (t < 128 ? dstA : dstB) + row * TSH;
  *(v8h*)(d)      = *(const v8h*)(s);
  *(v8h*)(d + 8)  = *(const v8h*)(s + 8);
  *(v8h*)(d + 16) = *(const v8h*)(s + 16);
  *(v8h*)(d + 24) = *(const v8h*)(s + 24);
}
#endif

// ---------------------------------------------------------------------------
// Weight transpose + f32 -> f16 convert:  W[K x N] -> Wt[N x K]
// ---------------------------------------------------------------------------
__global__ __launch_bounds__(256) void transpose_cvt(const float* __restrict__ W,
                                                     _Float16* __restrict__ Wt,
                                                     int K, int N) {
  __shared__ float tile[32][33];
  const int bx = blockIdx.x * 32;   // N
  const int by = blockIdx.y * 32;   // K
  const int tx = threadIdx.x, ty = threadIdx.y;  // 32 x 8
#pragma unroll
  for (int i = 0; i < 4; ++i) {
    int k = by + ty + 8 * i;
    tile[ty + 8 * i][tx] = W[(size_t)k * N + bx + tx];
  }
  __syncthreads();
#pragma unroll
  for (int i = 0; i < 4; ++i) {
    int n = bx + ty + 8 * i;
    Wt[(size_t)n * K + by + tx] = (_Float16)tile[tx][ty + 8 * i];
  }
}

// ---------------------------------------------------------------------------
// LayerNorm over D=1024, one row per 256-thread block, f16 output
// ---------------------------------------------------------------------------
__global__ __launch_bounds__(256) void ln_f16(const float* __restrict__ x,
                                              const float* __restrict__ g,
                                              const float* __restrict__ bb,
                                              _Float16* __restrict__ out) {
  __shared__ float sred[16];
  const int row = blockIdx.x;
  const float4 v = ((const float4*)(x + (size_t)row * Dc))[threadIdx.x];
  float s  = v.x + v.y + v.z + v.w;
  float ss = v.x * v.x + v.y * v.y + v.z * v.z + v.w * v.w;
#pragma unroll
  for (int off = 16; off; off >>= 1) {
    s  += __shfl_xor(s,  off, 32);
    ss += __shfl_xor(ss, off, 32);
  }
  const int w = threadIdx.x >> 5, lane = threadIdx.x & 31;
  if (lane == 0) { sred[w] = s; sred[8 + w] = ss; }
  __syncthreads();
  float ts = 0.f, tss = 0.f;
#pragma unroll
  for (int i = 0; i < 8; ++i) { ts += sred[i]; tss += sred[8 + i]; }
  const float mu   = ts * (1.0f / Dc);
  const float var  = tss * (1.0f / Dc) - mu * mu;
  const float rstd = rsqrtf(var + 1e-5f);
  const float4 gg = ((const float4*)g)[threadIdx.x];
  const float4 bv = ((const float4*)bb)[threadIdx.x];
  _Float16* o = out + (size_t)row * Dc + threadIdx.x * 4;
  o[0] = (_Float16)((v.x - mu) * rstd * gg.x + bv.x);
  o[1] = (_Float16)((v.y - mu) * rstd * gg.y + bv.y);
  o[2] = (_Float16)((v.z - mu) * rstd * gg.z + bv.z);
  o[3] = (_Float16)((v.w - mu) * rstd * gg.w + bv.w);
}

// ---------------------------------------------------------------------------
// WMMA GEMM with TDM double-buffered LDS staging.
// C[M x N] = A[M x K] (f16) * Bt[N x K]^T (f16) (+bias)(+resid)(gelu)
// 128x128 block tile, 8 waves, each wave 64x32 (4x2 tiles of 16x16).
// Per K-step: TDM DMAs next A/B 128x32 tiles into LDS while WMMAs consume the
// current ones from LDS (conflict-free padded layout).
// ---------------------------------------------------------------------------
template <int BIAS, int RESID, int GELU_, int OUTF16>
__global__ __launch_bounds__(256) void gemm_wmma(const _Float16* __restrict__ A,
                                                 const _Float16* __restrict__ Bt,
                                                 const float* __restrict__ bias,
                                                 const float* __restrict__ resid,
                                                 float* __restrict__ Cf,
                                                 _Float16* __restrict__ Ch,
                                                 int M, int N, int K) {
  __shared__ __align__(16) _Float16 smem[2][2][128 * TSH];  // 40 KB
  const int lane = threadIdx.x & 31, w = threadIdx.x >> 5;
  const int half = lane >> 4, r = lane & 15;
  const int wy = w >> 2, wx = w & 3;
  const int mblk = blockIdx.y * 128, nblk = blockIdx.x * 128;
  const _Float16* Ab = A + (size_t)mblk * K;
  const _Float16* Bb = Bt + (size_t)nblk * K;
  const int nsteps = K / 32;
  (void)M;
  v8f acc[4][2] = {};

#if HAVE_TDM
  if (w == 0) {
    tdm_load_tile32(Ab, K, lds_off(&smem[0][0][0]));
    tdm_load_tile32(Bb, K, lds_off(&smem[0][1][0]));
    wait_tensor0();
  }
  __syncthreads();
#else
  stage_sync(&smem[0][0][0], Ab, &smem[0][1][0], Bb, K);
  __syncthreads();
#endif

  for (int s = 0; s < nsteps; ++s) {
    const int cur = s & 1;
#if HAVE_TDM
    if ((w == 0) && (s + 1 < nsteps)) {
      tdm_load_tile32(Ab + (s + 1) * 32, K, lds_off(&smem[cur ^ 1][0][0]));
      tdm_load_tile32(Bb + (s + 1) * 32, K, lds_off(&smem[cur ^ 1][1][0]));
    }
#endif
    const _Float16* As = &smem[cur][0][0];
    const _Float16* Bs = &smem[cur][1][0];
    v16h bfr[2], afr[4];
#pragma unroll
    for (int j = 0; j < 2; ++j) bfr[j] = load_frag_lds(Bs, wx * 32 + 16 * j + r, half);
#pragma unroll
    for (int i = 0; i < 4; ++i) afr[i] = load_frag_lds(As, wy * 64 + 16 * i + r, half);
#pragma unroll
    for (int i = 0; i < 4; ++i)
#pragma unroll
      for (int j = 0; j < 2; ++j)
        acc[i][j] = __builtin_amdgcn_wmma_f32_16x16x32_f16(
            false, afr[i], false, bfr[j], (short)0, acc[i][j], false, false);
#if HAVE_TDM
    if ((w == 0) && (s + 1 < nsteps)) wait_tensor0();
    __syncthreads();
#else
    __syncthreads();
    if (s + 1 < nsteps) {
      stage_sync(&smem[cur ^ 1][0][0], Ab + (s + 1) * 32,
                 &smem[cur ^ 1][1][0], Bb + (s + 1) * 32, K);
      __syncthreads();
    }
#endif
  }

  const int m0 = mblk + wy * 64;
  const int n0 = nblk + wx * 32;
#pragma unroll
  for (int i = 0; i < 4; ++i)
#pragma unroll
    for (int j = 0; j < 2; ++j) {
#pragma unroll
      for (int vv = 0; vv < 8; ++vv) {
        const int m = m0 + 16 * i + vv + 8 * half;
        const int n = n0 + 16 * j + r;
        float x = acc[i][j][vv];
        if (BIAS)  x += bias[n];
        if (RESID) x += resid[(size_t)m * N + n];
        if (GELU_) x = 0.5f * x * (1.0f + erff(x * 0.70710678118654752f));
        if (OUTF16) Ch[(size_t)m * N + n] = (_Float16)x;
        else        Cf[(size_t)m * N + n] = x;
      }
    }
}

// ---------------------------------------------------------------------------
// RoPE + repack q,k: f32 [B,S,H,HD] -> f16 [B,H,S,HD]
// ---------------------------------------------------------------------------
__global__ __launch_bounds__(256) void rope_pack(const float* __restrict__ qf,
                                                 const float* __restrict__ kf,
                                                 const float* __restrict__ cs,
                                                 const float* __restrict__ sn,
                                                 _Float16* __restrict__ qh,
                                                 _Float16* __restrict__ kh) {
  const int idx = blockIdx.x * blockDim.x + threadIdx.x;  // B*S*H*HD/2 = 2^21
  const int d2 = idx & 31;
  const int h  = (idx >> 5) & 15;
  const int s  = (idx >> 9) & 2047;
  const int b  = idx >> 20;
  const float c  = cs[s * 32 + d2];
  const float si = sn[s * 32 + d2];
  const size_t in  = ((size_t)(b * Sc + s)) * Dc + h * HDc + 2 * d2;
  const size_t oot = (((size_t)(b * Hc + h)) * Sc + s) * HDc + 2 * d2;
  {
    const float xr = qf[in], xi = qf[in + 1];
    qh[oot]     = (_Float16)(xr * c - xi * si);
    qh[oot + 1] = (_Float16)(xr * si + xi * c);
  }
  {
    const float xr = kf[in], xi = kf[in + 1];
    kh[oot]     = (_Float16)(xr * c - xi * si);
    kh[oot + 1] = (_Float16)(xr * si + xi * c);
  }
}

// v: f32 [B,S,H,HD] -> f16 transposed [B,H,HD,S]
__global__ __launch_bounds__(256) void vpack(const float* __restrict__ vf,
                                             _Float16* __restrict__ vT) {
  const int idx = blockIdx.x * blockDim.x + threadIdx.x;  // B*S*D = 2^22
  const int d = idx & 63;
  const int h = (idx >> 6) & 15;
  const int s = (idx >> 10) & 2047;
  const int b = idx >> 21;
  vT[(((size_t)(b * Hc + h)) * HDc + d) * Sc + s] = (_Float16)vf[idx];
}

// ---------------------------------------------------------------------------
// Flash attention, one (b,h,16-q-rows) tile per wave.
// scores computed transposed (K·Q^T) so exp(scores) lands directly in the
// A-operand layout of the P·V WMMA — no cross-lane shuffles for P.
// ---------------------------------------------------------------------------
__global__ __launch_bounds__(256) void flash_attn(const _Float16* __restrict__ qh,
                                                  const _Float16* __restrict__ kh,
                                                  const _Float16* __restrict__ vT,
                                                  _Float16* __restrict__ ctx) {
  const int lane = threadIdx.x & 31, w = threadIdx.x >> 5;
  const int half = lane >> 4, r = lane & 15;
  const int tile = blockIdx.x * 8 + w;        // 4096 tiles total
  const int qt = tile & (Sc / 16 - 1);        // 0..127
  const int bh = tile >> 7;                   // 0..31
  const int b = bh >> 4, h = bh & 15;
  const _Float16* Q  = qh + (size_t)bh * Sc * HDc;
  const _Float16* Km = kh + (size_t)bh * Sc * HDc;
  const _Float16* Vt = vT + (size_t)bh * HDc * Sc;
  const int q0 = qt * 16;
  const float scale = 0.125f;  // 1/sqrt(64)

  v16h qb[2];
#pragma unroll
  for (int j = 0; j < 2; ++j) qb[j] = load_frag(Q, q0 + r, HDc, 32 * j, half);

  v8f o[4] = {};
  float mrow = -3.0e38f, lrow = 0.0f;

  for (int kt = 0; kt < Sc; kt += 32) {
    v8f s0 = {}, s1 = {};
#pragma unroll
    for (int j = 0; j < 2; ++j) {
      v16h a0 = load_frag(Km, kt + r, HDc, 32 * j, half);
      s0 = __builtin_amdgcn_wmma_f32_16x16x32_f16(false, a0, false, qb[j],
                                                  (short)0, s0, false, false);
      v16h a1 = load_frag(Km, kt + 16 + r, HDc, 32 * j, half);
      s1 = __builtin_amdgcn_wmma_f32_16x16x32_f16(false, a1, false, qb[j],
                                                  (short)0, s1, false, false);
    }
    // online softmax; per-lane stats are for q = lane & 15
    float mr = s0[0];
#pragma unroll
    for (int vv = 1; vv < 8; ++vv) mr = fmaxf(mr, s0[vv]);
#pragma unroll
    for (int vv = 0; vv < 8; ++vv) mr = fmaxf(mr, s1[vv]);
    mr = fmaxf(mr, __shfl_xor(mr, 16, 32));
    const float m_new = fmaxf(mrow, mr * scale);
    const float alpha = __expf(mrow - m_new);
    v16h pf;
    float psum = 0.0f;
#pragma unroll
    for (int vv = 0; vv < 8; ++vv) {
      const float p0 = __expf(s0[vv] * scale - m_new);
      const float p1 = __expf(s1[vv] * scale - m_new);
      pf[vv]     = (_Float16)p0;
      pf[vv + 8] = (_Float16)p1;
      psum += p0 + p1;
    }
    psum += __shfl_xor(psum, 16, 32);
    lrow = lrow * alpha + psum;
    mrow = m_new;
    // rescale running output: accumulator vgpr j belongs to q = j + 8*half
#pragma unroll
    for (int j = 0; j < 8; ++j) {
      const float aj = __shfl(alpha, j + 8 * half, 32);
#pragma unroll
      for (int t = 0; t < 4; ++t) o[t][j] *= aj;
    }
#pragma unroll
    for (int t = 0; t < 4; ++t) {
      v16h vfr = load_frag(Vt, 16 * t + r, Sc, kt, half);
      o[t] = __builtin_amdgcn_wmma_f32_16x16x32_f16(false, pf, false, vfr,
                                                    (short)0, o[t], false, false);
    }
  }
  const float linv = 1.0f / lrow;
#pragma unroll
  for (int j = 0; j < 8; ++j) {
    const float lj = __shfl(linv, j + 8 * half, 32);
#pragma unroll
    for (int t = 0; t < 4; ++t) {
      const float val = o[t][j] * lj;
      const size_t row = (size_t)(b * Sc + q0 + j + 8 * half);
      const int col = h * HDc + 16 * t + r;
      ctx[row * Dc + col] = (_Float16)val;
    }
  }
}

// ---------------------------------------------------------------------------
extern "C" void kernel_launch(void* const* d_in, const int* in_sizes, int n_in,
                              void* d_out, int out_size, void* d_ws, size_t ws_size,
                              hipStream_t stream) {
  (void)in_sizes; (void)n_in; (void)out_size; (void)ws_size;
  const float* x   = (const float*)d_in[0];
  const float* fc  = (const float*)d_in[2];
  const float* fs  = (const float*)d_in[3];
  const float* Wq  = (const float*)d_in[4];
  const float* Wk  = (const float*)d_in[5];
  const float* Wv  = (const float*)d_in[6];
  const float* Wo  = (const float*)d_in[7];
  const float* bo  = (const float*)d_in[8];
  const float* g1  = (const float*)d_in[9];
  const float* be1 = (const float*)d_in[10];
  const float* g2  = (const float*)d_in[11];
  const float* be2 = (const float*)d_in[12];
  const float* W1  = (const float*)d_in[13];
  const float* b1  = (const float*)d_in[14];
  const float* W2  = (const float*)d_in[15];
  const float* b2  = (const float*)d_in[16];
  float* outp = (float*)d_out;
  char* ws = (char*)d_ws;

  // workspace layout (with dead-buffer aliasing)
  size_t off = 0;
  const size_t XNH = off; off += (size_t)Mrows * Dc * 2;          // 8 MB (alias: CTXH)
  const size_t WQT = off; off += (size_t)Dc * Dc * 2;             // 2 MB
  const size_t WKT = off; off += (size_t)Dc * Dc * 2;
  const size_t WVT = off; off += (size_t)Dc * Dc * 2;
  const size_t WOT = off; off += (size_t)Dc * Dc * 2;
  const size_t W1T = off; off += (size_t)DFFc * Dc * 2;           // 8 MB
  const size_t W2T = off; off += (size_t)Dc * DFFc * 2;           // 8 MB
  const size_t QF  = off; off += (size_t)Mrows * Dc * 4;          // 16 MB (alias: HH lo)
  const size_t KF  = off; off += (size_t)Mrows * Dc * 4;          // 16 MB (alias: HH hi)
  const size_t VF  = off; off += (size_t)Mrows * Dc * 4;          // 16 MB
  const size_t QH  = off; off += (size_t)Mrows * Dc * 2;          // 8 MB (alias: XN2H)
  const size_t KH  = off; off += (size_t)Mrows * Dc * 2;
  const size_t VTT = off; off += (size_t)Mrows * Dc * 2;
  const size_t X1  = off; off += (size_t)Mrows * Dc * 4;          // 16 MB
  const size_t CTXH = XNH;   // xn_h dead after QKV GEMMs
  const size_t XN2H = QH;    // q_h dead after attention
  const size_t HH   = QF;    // q_f32/k_f32 dead after rope (32 MB region)

  _Float16* xnh  = (_Float16*)(ws + XNH);
  _Float16* wqt  = (_Float16*)(ws + WQT);
  _Float16* wkt  = (_Float16*)(ws + WKT);
  _Float16* wvt  = (_Float16*)(ws + WVT);
  _Float16* wot  = (_Float16*)(ws + WOT);
  _Float16* w1t  = (_Float16*)(ws + W1T);
  _Float16* w2t  = (_Float16*)(ws + W2T);
  float*    qf   = (float*)(ws + QF);
  float*    kf   = (float*)(ws + KF);
  float*    vf   = (float*)(ws + VF);
  _Float16* qh   = (_Float16*)(ws + QH);
  _Float16* kh   = (_Float16*)(ws + KH);
  _Float16* vtt  = (_Float16*)(ws + VTT);
  float*    x1   = (float*)(ws + X1);
  _Float16* ctxh = (_Float16*)(ws + CTXH);
  _Float16* xn2h = (_Float16*)(ws + XN2H);
  _Float16* hh   = (_Float16*)(ws + HH);

  const dim3 T256(256);
  const dim3 TT(32, 8);

  // 1) weights -> f16 transposed
  transpose_cvt<<<dim3(Dc / 32, Dc / 32), TT, 0, stream>>>(Wq, wqt, Dc, Dc);
  transpose_cvt<<<dim3(Dc / 32, Dc / 32), TT, 0, stream>>>(Wk, wkt, Dc, Dc);
  transpose_cvt<<<dim3(Dc / 32, Dc / 32), TT, 0, stream>>>(Wv, wvt, Dc, Dc);
  transpose_cvt<<<dim3(Dc / 32, Dc / 32), TT, 0, stream>>>(Wo, wot, Dc, Dc);
  transpose_cvt<<<dim3(DFFc / 32, Dc / 32), TT, 0, stream>>>(W1, w1t, Dc, DFFc);
  transpose_cvt<<<dim3(Dc / 32, DFFc / 32), TT, 0, stream>>>(W2, w2t, DFFc, Dc);

  // 2) LN1
  ln_f16<<<Mrows, T256, 0, stream>>>(x, g1, be1, xnh);

  // 3) QKV projections (f32 out)
  const dim3 Gqkv(Dc / 128, Mrows / 128);
  gemm_wmma<0, 0, 0, 0><<<Gqkv, T256, 0, stream>>>(xnh, wqt, nullptr, nullptr, qf, nullptr, Mrows, Dc, Dc);
  gemm_wmma<0, 0, 0, 0><<<Gqkv, T256, 0, stream>>>(xnh, wkt, nullptr, nullptr, kf, nullptr, Mrows, Dc, Dc);
  gemm_wmma<0, 0, 0, 0><<<Gqkv, T256, 0, stream>>>(xnh, wvt, nullptr, nullptr, vf, nullptr, Mrows, Dc, Dc);

  // 4) RoPE + head-major repack, V transpose-pack
  rope_pack<<<(Bc * Sc * Hc * (HDc / 2)) / 256, T256, 0, stream>>>(qf, kf, fc, fs, qh, kh);
  vpack<<<(Mrows * Dc) / 256, T256, 0, stream>>>(vf, vtt);

  // 5) flash attention
  flash_attn<<<(Bc * Hc * (Sc / 16)) / 8, T256, 0, stream>>>(qh, kh, vtt, ctxh);

  // 6) output projection + bias + residual (f32)
  gemm_wmma<1, 1, 0, 0><<<Gqkv, T256, 0, stream>>>(ctxh, wot, bo, x, x1, nullptr, Mrows, Dc, Dc);

  // 7) LN2
  ln_f16<<<Mrows, T256, 0, stream>>>(x1, g2, be2, xn2h);

  // 8) MLP up + exact GELU (f16 out)
  gemm_wmma<1, 0, 1, 1><<<dim3(DFFc / 128, Mrows / 128), T256, 0, stream>>>(
      xn2h, w1t, b1, nullptr, nullptr, hh, Mrows, DFFc, Dc);

  // 9) MLP down + bias + residual -> d_out (f32)
  gemm_wmma<1, 1, 0, 0><<<Gqkv, T256, 0, stream>>>(hh, w2t, b2, x1, outp, nullptr, Mrows, Dc, DFFc);
}
